// ProbabilityMatrix_31885837205965
// MI455X (gfx1250) — matrix operand order repaired and checked by reading, not compiled, not run on hardware
//
#include <hip/hip_runtime.h>
#include <hip/hip_bf16.h>

// CDNA5 / gfx1250: wave32, WMMA 16x16x32 f16->f32.
typedef __attribute__((ext_vector_type(16))) _Float16 v16h;
typedef __attribute__((ext_vector_type(8)))  float    v8f;
typedef __attribute__((ext_vector_type(4)))  float    v4f;

#define BATCH  16
#define PPB    32768                 // patches per batch row
#define NPATCH (BATCH * PPB)         // 524288 patches per tensor

// ---------------------------------------------------------------------------
// Zero the integer count workspace (graph-capture safe; no hipMemset needed).
// ---------------------------------------------------------------------------
__global__ void pm_zero_ws(int* __restrict__ p, int n) {
  int i = blockIdx.x * blockDim.x + threadIdx.x;
  if (i < n) p[i] = 0;
}

// ---------------------------------------------------------------------------
// s = 4 tensors: 16 floats per patch. One WMMA (16x16x32 f16) computes the
// sums of 32 patches: A row m packs patch 2m (K 0..15) and patch 2m+1
// (K 16..31); B col0 = (k<16), col1 = (k>=16). Binary data -> f16 conversion
// is exact; accumulation is f32 -> exact integer sums.
//
// A-fragment layout (ISA 7.12.2, 16-bit A 16x32):
//   lanes 0-15 (row m=lane):  VGPR0-3 = K0-7,  VGPR4-7 = K16-23
//   lanes 16-31 (row m=lane-16): VGPR0-3 = K8-15, VGPR4-7 = K24-31
// B-fragment layout matches A (col n = lane%16): elems 0-7 = patch 2m K,
// elems 8-15 = patch 2m+1 K -> col0 ones in elems 0-7, col1 ones in elems 8-15.
// D layout: lane L<16: VGPR r = D[r][L]; lane L>=16: VGPR r = D[8+r][L-16].
// => patch sums land in lanes {0,1,16,17}, 8 each.
// ---------------------------------------------------------------------------
__global__ void pm_hist_s4(const float* __restrict__ src, int* __restrict__ gcounts) {
  __shared__ int lhist[BATCH * 16];
  for (int i = threadIdx.x; i < BATCH * 16; i += blockDim.x) lhist[i] = 0;
  __syncthreads();

  const int lane          = threadIdx.x & 31;
  const int wavesPerBlock = blockDim.x >> 5;
  const int gwave         = blockIdx.x * wavesPerBlock + (threadIdx.x >> 5);
  const int nwaves        = gridDim.x * wavesPerBlock;
  const int ntiles        = NPATCH / 32;      // 32 patches per tile

  v16h bfrag;
  {
    #pragma unroll
    for (int i = 0; i < 16; ++i)
      bfrag[i] = (_Float16)(((lane & 15) == (i >> 3)) ? 1.0f : 0.0f);
  }

  const int m    = lane & 15;
  const int half = lane >> 4;   // 0: elems 0-7 of each patch, 1: elems 8-15

  for (int tile = gwave; tile < ntiles; tile += nwaves) {   // uniform per wave
    const size_t base = (size_t)tile * (32 * 16);
    const int    b    = (tile * 32) / PPB;                  // PPB % 32 == 0

    const float* pa = src + base + (size_t)(2 * m)     * 16 + half * 8;
    const float* pb = src + base + (size_t)(2 * m + 1) * 16 + half * 8;
    v4f a0 = *(const v4f*)(pa);
    v4f a1 = *(const v4f*)(pa + 4);
    v4f a2 = *(const v4f*)(pb);
    v4f a3 = *(const v4f*)(pb + 4);

    v16h A;
    #pragma unroll
    for (int i = 0; i < 4; ++i) {
      A[i]      = (_Float16)a0[i];
      A[4 + i]  = (_Float16)a1[i];
      A[8 + i]  = (_Float16)a2[i];
      A[12 + i] = (_Float16)a3[i];
    }

    v8f c = {};
    c = __builtin_amdgcn_wmma_f32_16x16x32_f16(false, A, false, bfrag,
                                               (short)0, c, false, false);

    if (lane == 0 || lane == 1 || lane == 16 || lane == 17) {
      #pragma unroll
      for (int r = 0; r < 8; ++r) {
        int v = (int)(c[r] + 0.5f);
        if (v < 16) atomicAdd(&lhist[b * 16 + v], 1);   // drop v == s^2
      }
    }
  }

  __syncthreads();
  for (int i = threadIdx.x; i < BATCH * 16; i += blockDim.x)
    if (lhist[i]) atomicAdd(&gcounts[i], lhist[i]);
}

// ---------------------------------------------------------------------------
// s = 8 tensors: 64 floats per patch. Two K=32 WMMAs accumulate one patch
// per A row (elems 0..31 then 32..63); B col0 = all ones. 16 patch sums per
// tile, landing in lanes {0,16}, 8 each.
// ---------------------------------------------------------------------------
__global__ void pm_hist_s8(const float* __restrict__ src, int* __restrict__ gcounts) {
  __shared__ int lhist[BATCH * 64];
  for (int i = threadIdx.x; i < BATCH * 64; i += blockDim.x) lhist[i] = 0;
  __syncthreads();

  const int lane          = threadIdx.x & 31;
  const int wavesPerBlock = blockDim.x >> 5;
  const int gwave         = blockIdx.x * wavesPerBlock + (threadIdx.x >> 5);
  const int nwaves        = gridDim.x * wavesPerBlock;
  const int ntiles        = NPATCH / 16;      // 16 patches per tile

  v16h bfrag;   // B col 0 = all ones (lane0: K0-7,16-23, lane16: K8-15,24-31)
  {
    _Float16 bv = (_Float16)((lane == 0 || lane == 16) ? 1.0f : 0.0f);
    #pragma unroll
    for (int i = 0; i < 16; ++i) bfrag[i] = bv;
  }

  const int m  = lane & 15;
  const int eb = (lane >> 4) * 8;   // K-subrange this lane supplies

  for (int tile = gwave; tile < ntiles; tile += nwaves) {
    const size_t base = (size_t)tile * (16 * 64);
    const int    b    = (tile * 16) / PPB;                  // PPB % 16 == 0
    const float* pr   = src + base + (size_t)m * 64;

    v8f c = {};
    #pragma unroll
    for (int hfm = 0; hfm < 2; ++hfm) {       // K 0..31 then K 32..63
      const int e0 = hfm * 32 + eb;           // elems for VGPR0-3 (lo K group)
      const int e1 = hfm * 32 + 16 + eb;      // elems for VGPR4-7 (hi K group)
      v4f a0 = *(const v4f*)(pr + e0);
      v4f a1 = *(const v4f*)(pr + e0 + 4);
      v4f a2 = *(const v4f*)(pr + e1);
      v4f a3 = *(const v4f*)(pr + e1 + 4);
      v16h A;
      #pragma unroll
      for (int i = 0; i < 4; ++i) {
        A[i]      = (_Float16)a0[i];
        A[4 + i]  = (_Float16)a1[i];
        A[8 + i]  = (_Float16)a2[i];
        A[12 + i] = (_Float16)a3[i];
      }
      c = __builtin_amdgcn_wmma_f32_16x16x32_f16(false, A, false, bfrag,
                                                 (short)0, c, false, false);
    }

    if (lane == 0 || lane == 16) {
      #pragma unroll
      for (int r = 0; r < 8; ++r) {
        int v = (int)(c[r] + 0.5f);
        if (v < 64) atomicAdd(&lhist[b * 64 + v], 1);   // drop v == s^2
      }
    }
  }

  __syncthreads();
  for (int i = threadIdx.x; i < BATCH * 64; i += blockDim.x)
    if (lhist[i]) atomicAdd(&gcounts[i], lhist[i]);
}

// ---------------------------------------------------------------------------
// Row-normalize counts -> probabilities. probs = (counts/P) / sum(counts/P)
// == counts / rowsum(counts).  One block per batch row, 64 threads.
// ---------------------------------------------------------------------------
__global__ void pm_finalize(const int* __restrict__ counts, float* __restrict__ out,
                            int bins) {
  __shared__ float ssum[64];
  const int row = blockIdx.x;
  const int t   = threadIdx.x;
  float v = (t < bins) ? (float)counts[row * bins + t] : 0.0f;
  ssum[t] = v;
  __syncthreads();
  #pragma unroll
  for (int off = 32; off > 0; off >>= 1) {
    if (t < off) ssum[t] += ssum[t + off];
    __syncthreads();
  }
  const float total = ssum[0];
  if (t < bins) out[row * bins + t] = v / total;
}

// ---------------------------------------------------------------------------
extern "C" void kernel_launch(void* const* d_in, const int* in_sizes, int n_in,
                              void* d_out, int out_size, void* d_ws, size_t ws_size,
                              hipStream_t stream) {
  (void)in_sizes; (void)n_in; (void)out_size; (void)ws_size;

  const float* cd0_box0 = (const float*)d_in[0];  // [16,32768,4,4]
  const float* cd0_box1 = (const float*)d_in[1];  // [16,32768,8,8]
  const float* cd1_box0 = (const float*)d_in[2];  // [16,32768,4,4]
  const float* cd1_box1 = (const float*)d_in[3];  // [16,32768,8,8]
  float* out = (float*)d_out;

  // Output / count segment offsets (elements): [16,16],[16,64],[16,16],[16,64]
  const int off0 = 0, off1 = 256, off2 = 1280, off3 = 1536;
  const int total_counts = 2560;
  int* counts = (int*)d_ws;

  pm_zero_ws<<<(total_counts + 255) / 256, 256, 0, stream>>>(counts, total_counts);

  pm_hist_s4<<<1024, 256, 0, stream>>>(cd0_box0, counts + off0);
  pm_hist_s8<<<1024, 256, 0, stream>>>(cd0_box1, counts + off1);
  pm_hist_s4<<<1024, 256, 0, stream>>>(cd1_box0, counts + off2);
  pm_hist_s8<<<1024, 256, 0, stream>>>(cd1_box1, counts + off3);

  pm_finalize<<<BATCH, 64, 0, stream>>>(counts + off0, out + off0, 16);
  pm_finalize<<<BATCH, 64, 0, stream>>>(counts + off1, out + off1, 64);
  pm_finalize<<<BATCH, 64, 0, stream>>>(counts + off2, out + off2, 16);
  pm_finalize<<<BATCH, 64, 0, stream>>>(counts + off3, out + off3, 64);
}
